// MultiHeadAttention_21680994910837
// MI455X (gfx1250) — matrix operand order repaired and checked by reading, not compiled, not run on hardware
//
#include <hip/hip_runtime.h>

// ---------------------------------------------------------------- constants
#define D_MODEL 1024
#define NUM_HEADS 16
#define DEPTH 64
#define BATCH 2
#define SEQ 2048
#define BS (BATCH * SEQ)          // 4096 rows
#define DD (D_MODEL * D_MODEL)    // weight elements
#define BSD (BS * D_MODEL)        // activation elements

// ---------------------------------------------------------------- types
typedef __bf16 bf16;
typedef __attribute__((ext_vector_type(16))) __bf16 bf16x16;
typedef __attribute__((ext_vector_type(8)))  __bf16 bf16x8;
typedef __attribute__((ext_vector_type(8)))  float  f32x8;
typedef __attribute__((ext_vector_type(4)))  float  f32x4;

// ---------------------------------------------------------------- helpers
__device__ __forceinline__ bf16 f2bf(float x) {
  union { float f; unsigned u; } v; v.f = x;
  unsigned r = v.u + 0x7FFFu + ((v.u >> 16) & 1u);   // round-to-nearest-even
  unsigned short h = (unsigned short)(r >> 16);
  bf16 out;
  __builtin_memcpy(&out, &h, sizeof(h));
  return out;
}

// hardware exp2 (v_exp_f32)
#if __has_builtin(__builtin_amdgcn_exp2f)
__device__ __forceinline__ float hw_exp2(float x) {
  return __builtin_amdgcn_exp2f(x);
}
#else
__device__ __forceinline__ float hw_exp2(float x) { return exp2f(x); }
#endif

__device__ __forceinline__ f32x8 zero8() {
  f32x8 z;
#pragma unroll
  for (int i = 0; i < 8; ++i) z[i] = 0.0f;
  return z;
}

// ---- 16-lane-group reductions.  DPP16 rows are 16 lanes on wave32, so
// quad_perm xor1/xor2 + row_half_mirror + row_mirror reduce within each
// 16-lane half without touching LDS (vs ds_bpermute + s_wait_dscnt).
#if __has_builtin(__builtin_amdgcn_update_dpp)
#define DPPF(x, ctrl)                                                         \
  __builtin_bit_cast(float, __builtin_amdgcn_update_dpp(                      \
                                0, __builtin_bit_cast(int, (x)), (ctrl), 0xF, \
                                0xF, true))
__device__ __forceinline__ float red_max16(float x) {
  x = fmaxf(x, DPPF(x, 0xB1));   // quad_perm(1,0,3,2)  : xor 1
  x = fmaxf(x, DPPF(x, 0x4E));   // quad_perm(2,3,0,1)  : xor 2
  x = fmaxf(x, DPPF(x, 0x141));  // row_half_mirror     : xor 7 (crosses quads)
  x = fmaxf(x, DPPF(x, 0x140));  // row_mirror          : xor 15 (crosses 8s)
  return x;
}
__device__ __forceinline__ float red_sum16(float x) {
  x += DPPF(x, 0xB1);
  x += DPPF(x, 0x4E);
  x += DPPF(x, 0x141);
  x += DPPF(x, 0x140);
  return x;
}
#else
__device__ __forceinline__ float red_max16(float x) {
#pragma unroll
  for (int m = 1; m < 16; m <<= 1) x = fmaxf(x, __shfl_xor(x, m, 32));
  return x;
}
__device__ __forceinline__ float red_sum16(float x) {
#pragma unroll
  for (int m = 1; m < 16; m <<= 1) x += __shfl_xor(x, m, 32);
  return x;
}
#endif

// A-fragment (16x32 bf16, k = 16*half + s at call sites): two contiguous 8-element chunks / lane
__device__ __forceinline__ bf16x16 load_a(const bf16* p0, const bf16* p1) {
  union { bf16x16 v; bf16x8 h[2]; } u;
  u.h[0] = *(const bf16x8*)p0;
  u.h[1] = *(const bf16x8*)p1;
  return u.v;
}
// Same, but source is fp32 (convert in-register)
__device__ __forceinline__ bf16x16 load_a(const float* p0, const float* p1) {
  f32x4 a0 = *(const f32x4*)(p0);
  f32x4 a1 = *(const f32x4*)(p0 + 4);
  f32x4 b0 = *(const f32x4*)(p1);
  f32x4 b1 = *(const f32x4*)(p1 + 4);
  bf16x16 v;
#pragma unroll
  for (int i = 0; i < 4; ++i) {
    v[i]      = f2bf(a0[i]);
    v[4 + i]  = f2bf(a1[i]);
    v[8 + i]  = f2bf(b0[i]);
    v[12 + i] = f2bf(b1[i]);
  }
  return v;
}

__device__ __forceinline__ f32x8 wmma_bf16(bf16x16 a, bf16x16 b, f32x8 c) {
  return __builtin_amdgcn_wmma_f32_16x16x32_bf16(
      /*neg_a=*/false, a, /*neg_b=*/false, b,
      /*c_mod=*/(short)0, c, /*reuse_a=*/false, /*reuse_b=*/false);
}

// ---------------------------------------------------------------- kernel 1:
// W (K x N, f32, row-major) -> Wt (N x K, bf16) via LDS tile transpose
__global__ void transpose_conv_kernel(const float* __restrict__ W,
                                      bf16* __restrict__ Wt,
                                      int Kd, int Nd) {
  __shared__ float tile[32][33];
  const int k0 = blockIdx.y * 32;
  const int n0 = blockIdx.x * 32;
  const int tx = threadIdx.x;  // 0..31
  const int ty = threadIdx.y;  // 0..7
#pragma unroll
  for (int i = 0; i < 32; i += 8)
    tile[ty + i][tx] = W[(size_t)(k0 + ty + i) * Nd + (n0 + tx)];
  __syncthreads();
#pragma unroll
  for (int i = 0; i < 32; i += 8)
    Wt[(size_t)(n0 + ty + i) * Kd + (k0 + tx)] = f2bf(tile[tx][ty + i]);
}

// ---------------------------------------------------------------- kernel 2:
// C[M,N] = A[M,K] @ Bt[N,K]^T + bias.  256 thr = 8 waves; wave = 16x64 strip.
template <typename AT, bool OUT_BF16>
__global__ void gemm_wmma_kernel(const AT* __restrict__ A,
                                 const bf16* __restrict__ Bt,
                                 const float* __restrict__ bias,
                                 void* __restrict__ Cout,
                                 int M, int N, int Kd) {
  const int lane = threadIdx.x & 31;
  const int w    = threadIdx.x >> 5;
  const int half = lane >> 4;     // K-group for A/B fragments
  const int nl   = lane & 15;     // M index (A) / N index (B,C)
  const int mi   = w & 1;         // 2 M-tiles per WG
  const int ng   = w >> 1;        // 4 N-groups of 64 per WG
  const int mrowA = blockIdx.y * 32 + mi * 16 + nl;
  const int nbase = blockIdx.x * 256 + ng * 64;

  f32x8 c[4];
#pragma unroll
  for (int t = 0; t < 4; ++t) c[t] = zero8();

  const AT* arow = A + (size_t)mrowA * Kd;
  for (int k0 = 0; k0 < Kd; k0 += 32) {
    __builtin_prefetch(arow + k0 + 256, 0, 1);
    bf16x16 afrag = load_a(arow + k0 + 16 * half, arow + k0 + 16 * half + 8);
#pragma unroll
    for (int t = 0; t < 4; ++t) {
      const bf16* bp = Bt + (size_t)(nbase + t * 16 + nl) * Kd + k0 + 16 * half;
      bf16x16 bfrag = *(const bf16x16*)bp;
      c[t] = wmma_bf16(afrag, bfrag, c[t]);
    }
  }

#pragma unroll
  for (int t = 0; t < 4; ++t) {
    const int ncol = nbase + t * 16 + nl;
    const float bv = bias[ncol];
#pragma unroll
    for (int r = 0; r < 8; ++r) {
      const int mrow = blockIdx.y * 32 + mi * 16 + r + 8 * half;
      const float val = c[t][r] + bv;
      if (OUT_BF16)
        ((bf16*)Cout)[(size_t)mrow * N + ncol] = f2bf(val);
      else
        ((float*)Cout)[(size_t)mrow * N + ncol] = val;
    }
  }
}

// ---------------------------------------------------------------- kernel 3:
// Flash attention: 8 waves/WG, each wave owns a 16-query tile of one (b,h).
// Keys streamed in tiles of 64; V staged transposed in wave-private LDS.
// Softmax runs in exp2 domain (hardware transcendental is exp2).
__global__ void flash_attn_kernel(const bf16* __restrict__ Qp,
                                  const bf16* __restrict__ Kp,
                                  const bf16* __restrict__ Vp,
                                  bf16* __restrict__ Ctx) {
  // per-wave: 64x64 transposed V tile (4096) + 16x64 P tile (1024)
  __shared__ __align__(32) bf16 smem[8 * 5120];

  const int lane = threadIdx.x & 31;
  const int w    = threadIdx.x >> 5;
  const int half = lane >> 4;
  const int nl   = lane & 15;

  const int idx  = blockIdx.x;        // b*256 + h*16 + qblk
  const int qblk = idx & 15;
  const int h    = (idx >> 4) & 15;
  const int b    = idx >> 8;
  const int s0   = qblk * 128 + w * 16;

  bf16* vt = smem + w * 5120;         // [depth 0..63][key 0..63]
  bf16* pt = vt + 4096;               // [row 0..15][key 0..63]

  // Q A-fragments (depth 0..31 and 32..63), row = nl
  const bf16* qrow = Qp + (size_t)(b * SEQ + s0 + nl) * D_MODEL + h * DEPTH;
  const bf16x16 qa0 = load_a(qrow + 16 * half,      qrow + 8 + 16 * half);
  const bf16x16 qa1 = load_a(qrow + 32 + 16 * half, qrow + 40 + 16 * half);

  float mrow[8], lrow[8], alr[8];
  f32x8 acc[4];
#pragma unroll
  for (int r = 0; r < 8; ++r) { mrow[r] = -1e30f; lrow[r] = 0.0f; }
#pragma unroll
  for (int t = 0; t < 4; ++t) acc[t] = zero8();

  // scores scaled into exp2 domain: x = score * (1/sqrt(64)) * log2(e)
  const float scale2 = 0.125f * 1.44269504f;

  for (int kt = 0; kt < SEQ / 64; ++kt) {
    const int kbase = kt * 64;

    // ---- stage 64x64 V tile transposed; each lane owns 2 key rows
    {
      const bf16* vr0 =
          Vp + (size_t)(b * SEQ + kbase + lane) * D_MODEL + h * DEPTH;
      const bf16* vr1 = vr0 + (size_t)32 * D_MODEL;
#pragma unroll
      for (int c0 = 0; c0 < 64; c0 += 8) {
        bf16x8 ch0 = *(const bf16x8*)(vr0 + c0);
        bf16x8 ch1 = *(const bf16x8*)(vr1 + c0);
#pragma unroll
        for (int e = 0; e < 8; ++e) {
          vt[(c0 + e) * 64 + lane]      = ch0[e];
          vt[(c0 + e) * 64 + 32 + lane] = ch1[e];
        }
      }
    }

    // ---- scores: four 16x16 N-tiles, K=depth in two 32-steps each
    f32x8 s[4];
#pragma unroll
    for (int j = 0; j < 4; ++j) {
      const bf16* kr =
          Kp + (size_t)(b * SEQ + kbase + j * 16 + nl) * D_MODEL + h * DEPTH;
      bf16x16 kb0 = *(const bf16x16*)(kr + 16 * half);
      bf16x16 kb1 = *(const bf16x16*)(kr + 32 + 16 * half);
      f32x8 z = zero8();
      z = wmma_bf16(qa0, kb0, z);
      z = wmma_bf16(qa1, kb1, z);
      s[j] = z;
    }

    // ---- online softmax update (row = r + 8*half, N along 16-lane group)
#pragma unroll
    for (int r = 0; r < 8; ++r) {
      float x0 = s[0][r] * scale2;
      float x1 = s[1][r] * scale2;
      float x2 = s[2][r] * scale2;
      float x3 = s[3][r] * scale2;
      float v = fmaxf(fmaxf(x0, x1), fmaxf(x2, x3));
      v = red_max16(v);
      const float mnew = fmaxf(mrow[r], v);
      const float al = hw_exp2(mrow[r] - mnew);
      const float p0 = hw_exp2(x0 - mnew);
      const float p1 = hw_exp2(x1 - mnew);
      const float p2 = hw_exp2(x2 - mnew);
      const float p3 = hw_exp2(x3 - mnew);
      float ps = (p0 + p1) + (p2 + p3);
      ps = red_sum16(ps);
      lrow[r] = lrow[r] * al + ps;
      mrow[r] = mnew;
      alr[r]  = al;
      const int rowg = r + 8 * half;
      pt[rowg * 64 + nl]      = f2bf(p0);
      pt[rowg * 64 + 16 + nl] = f2bf(p1);
      pt[rowg * 64 + 32 + nl] = f2bf(p2);
      pt[rowg * 64 + 48 + nl] = f2bf(p3);
    }

    // ---- P (16x64) as two A-fragments from LDS (keys 0..31 and 32..63)
    bf16x16 pa0 =
        load_a(pt + nl * 64 + 16 * half, pt + nl * 64 + 8 + 16 * half);
    bf16x16 pa1 =
        load_a(pt + nl * 64 + 32 + 16 * half, pt + nl * 64 + 40 + 16 * half);

    // ---- O[t] = P @ Vt + alpha*O[t]   (two K=32 steps per output tile)
#pragma unroll
    for (int t = 0; t < 4; ++t) {
      const bf16* vrow = vt + (t * 16 + nl) * 64;
      bf16x16 vb0 = *(const bf16x16*)(vrow + 16 * half);
      bf16x16 vb1 = *(const bf16x16*)(vrow + 32 + 16 * half);
      f32x8 cm;
#pragma unroll
      for (int r = 0; r < 8; ++r) cm[r] = acc[t][r] * alr[r];
      f32x8 tmp = wmma_bf16(pa0, vb0, cm);
      acc[t] = wmma_bf16(pa1, vb1, tmp);
    }
  }

  // ---- epilogue: normalize and write context (bf16)
#pragma unroll
  for (int t = 0; t < 4; ++t) {
#pragma unroll
    for (int r = 0; r < 8; ++r) {
      const int row = s0 + r + 8 * half;
      const float o = acc[t][r] / lrow[r];
      Ctx[(size_t)(b * SEQ + row) * D_MODEL + h * DEPTH + t * 16 + nl] = f2bf(o);
    }
  }
}

// ---------------------------------------------------------------- launcher
extern "C" void kernel_launch(void* const* d_in, const int* in_sizes, int n_in,
                              void* d_out, int out_size, void* d_ws, size_t ws_size,
                              hipStream_t stream) {
  const float* v  = (const float*)d_in[0];
  const float* k  = (const float*)d_in[1];
  const float* q  = (const float*)d_in[2];
  const float* wq = (const float*)d_in[3];
  const float* bq = (const float*)d_in[4];
  const float* wk = (const float*)d_in[5];
  const float* bk = (const float*)d_in[6];
  const float* wv = (const float*)d_in[7];
  const float* bv = (const float*)d_in[8];
  const float* wo = (const float*)d_in[9];
  const float* bo = (const float*)d_in[10];

  bf16* ws  = (bf16*)d_ws;
  bf16* wqT = ws;                 // D x D (N-major)
  bf16* wkT = wqT + DD;
  bf16* wvT = wkT + DD;
  bf16* woT = wvT + DD;
  bf16* Qp  = woT + DD;           // BS x D
  bf16* Kp  = Qp + BSD;
  bf16* Vp  = Kp + BSD;
  bf16* Ctx = Vp + BSD;

  // 1) transpose+convert the four weight matrices
  {
    dim3 grid(D_MODEL / 32, D_MODEL / 32), block(32, 8);
    transpose_conv_kernel<<<grid, block, 0, stream>>>(wq, wqT, D_MODEL, D_MODEL);
    transpose_conv_kernel<<<grid, block, 0, stream>>>(wk, wkT, D_MODEL, D_MODEL);
    transpose_conv_kernel<<<grid, block, 0, stream>>>(wv, wvT, D_MODEL, D_MODEL);
    transpose_conv_kernel<<<grid, block, 0, stream>>>(wo, woT, D_MODEL, D_MODEL);
  }

  // 2) Q/K/V projections (fp32 activations -> bf16 WMMA -> bf16 out)
  {
    dim3 grid(D_MODEL / 256, BS / 32), block(256);
    gemm_wmma_kernel<float, true><<<grid, block, 0, stream>>>(
        q, wqT, bq, Qp, BS, D_MODEL, D_MODEL);
    gemm_wmma_kernel<float, true><<<grid, block, 0, stream>>>(
        k, wkT, bk, Kp, BS, D_MODEL, D_MODEL);
    gemm_wmma_kernel<float, true><<<grid, block, 0, stream>>>(
        v, wvT, bv, Vp, BS, D_MODEL, D_MODEL);
  }

  // 3) flash attention: grid = B * H * (S/128) workgroups, 8 waves each
  {
    dim3 grid(BATCH * NUM_HEADS * (SEQ / 128)), block(256);
    flash_attn_kernel<<<grid, block, 0, stream>>>(Qp, Kp, Vp, Ctx);
  }

  // 4) output projection (bf16 ctx @ woT + bo -> fp32 d_out)
  {
    dim3 grid(D_MODEL / 256, BS / 32), block(256);
    gemm_wmma_kernel<bf16, false><<<grid, block, 0, stream>>>(
        Ctx, woT, bo, d_out, BS, D_MODEL, D_MODEL);
  }
}
